// LinearAttention_70557722738927
// MI455X (gfx1250) — matrix operand order repaired and checked by reading, not compiled, not run on hardware
//
#include <hip/hip_runtime.h>

typedef __attribute__((ext_vector_type(16))) _Float16 v16h;
typedef __attribute__((ext_vector_type(8)))  _Float16 v8h;
typedef __attribute__((ext_vector_type(8)))  float    v8f;

#define B_    4
#define H_    16
#define S_    4096
#define D_    64
#define BH_   64
#define CHUNK 512
#define KSTEP 32
#define ROWS  128

union V16U { v16h v; v8h h[2]; };

__global__ void zero_ws_kernel(float* __restrict__ p, int n) {
  int i = blockIdx.x * 256 + threadIdx.x;
  if (i < n) p[i] = 0.f;
}

// Phase 1: attn_inter[bh] += K_chunk^T @ V_chunk (f16 WMMA, fp32 accum), ksum += sum_s K
__global__ void __launch_bounds__(256)
la_phase1(const float* __restrict__ K, const float* __restrict__ V,
          const int* __restrict__ mask, float* __restrict__ inter,
          float* __restrict__ ksum)
{
  __shared__ __align__(16) _Float16 kT[64][40];  // kT[d][s], stride 40 halves = 80B (16B aligned, bank-spread)
  __shared__ __align__(16) _Float16 vT[64][40];  // vT[e][s]

  const int bh   = blockIdx.x;
  const int b    = bh >> 4;                      // H_ == 16
  const int tid  = threadIdx.x;
  const int wave = tid >> 5;
  const int lane = tid & 31;
  const int mt   = wave >> 1;                    // output row-tile (d)
  const int ntb  = (wave & 1) * 2;               // output col-tile pair (e)

  const float* Kp = K + (size_t)bh * S_ * D_;
  const float* Vp = V + (size_t)bh * S_ * D_;
  const int*   mp = mask + b * S_;

  v8f c0 = {}; v8f c1 = {};
  // ksum partials: this thread's dcol group is (tid&15)*4 for BOTH load passes,
  // so 4 fp32 register accumulators cover it (fed straight from global loads).
  float ka0 = 0.f, ka1 = 0.f, ka2 = 0.f, ka3 = 0.f;

  const int am  = mt * 16 + (lane & 15);         // A row (d)
  const int akb = (lane < 16) ? 0 : 8;           // A k-base (ISA 16-bit A layout)
  const int bn0 = ntb * 16 + (lane & 15);        // B col (e), tile 0
  const int bn1 = bn0 + 16;                      // B col (e), tile 1
  const int bkb = (lane < 16) ? 0 : 8;           // B k-base (ISA 16-bit B layout)

  const int s_base = blockIdx.y * CHUNK;

  for (int step = 0; step < CHUNK / KSTEP; ++step) {
    const int s0 = s_base + step * KSTEP;
    __syncthreads();
    #pragma unroll
    for (int it = 0; it < 2; ++it) {
      const int f    = tid + it * 256;           // float4 index 0..511
      const int srow = f >> 4;                   // 0..31
      const int dcol = (f & 15) * 4;             // == (tid&15)*4, invariant over it
      const float msk = mp[s0 + srow] ? 1.f : 0.f;
      const float4 kq = *(const float4*)(Kp + (size_t)(s0 + srow) * D_ + dcol);
      const float k0 = fmaxf(kq.x, 0.f) * msk;
      const float k1 = fmaxf(kq.y, 0.f) * msk;
      const float k2 = fmaxf(kq.z, 0.f) * msk;
      const float k3 = fmaxf(kq.w, 0.f) * msk;
      ka0 += k0; ka1 += k1; ka2 += k2; ka3 += k3;
      kT[dcol + 0][srow] = (_Float16)k0;
      kT[dcol + 1][srow] = (_Float16)k1;
      kT[dcol + 2][srow] = (_Float16)k2;
      kT[dcol + 3][srow] = (_Float16)k3;
      const float4 vq = *(const float4*)(Vp + (size_t)(s0 + srow) * D_ + dcol);
      vT[dcol + 0][srow] = (_Float16)vq.x;
      vT[dcol + 1][srow] = (_Float16)vq.y;
      vT[dcol + 2][srow] = (_Float16)vq.z;
      vT[dcol + 3][srow] = (_Float16)vq.w;
    }
    __syncthreads();

    V16U a, b0, b1;
    a.h[0]  = *(const v8h*)&kT[am][akb];
    a.h[1]  = *(const v8h*)&kT[am][akb + 16];
    b0.h[0] = *(const v8h*)&vT[bn0][bkb];
    b0.h[1] = *(const v8h*)&vT[bn0][bkb + 16];
    b1.h[0] = *(const v8h*)&vT[bn1][bkb];
    b1.h[1] = *(const v8h*)&vT[bn1][bkb + 16];

    c0 = __builtin_amdgcn_wmma_f32_16x16x32_f16(false, a.v, false, b0.v, (short)0, c0, false, false);
    c1 = __builtin_amdgcn_wmma_f32_16x16x32_f16(false, a.v, false, b1.v, (short)0, c1, false, false);
  }

  float* ip = inter + (size_t)bh * 4096;
  const int rbase = mt * 16 + ((lane < 16) ? 0 : 8);  // C layout: row = vgpr + lane-half*8
  const int col0  = ntb * 16 + (lane & 15);
  #pragma unroll
  for (int r = 0; r < 8; ++r) {
    atomicAdd(&ip[(rbase + r) * 64 + col0],      c0[r]);
    atomicAdd(&ip[(rbase + r) * 64 + col0 + 16], c1[r]);
  }
  float* kp = ksum + bh * 64 + (tid & 15) * 4;
  atomicAdd(kp + 0, ka0);
  atomicAdd(kp + 1, ka1);
  atomicAdd(kp + 2, ka2);
  atomicAdd(kp + 3, ka3);
}

// Phase 2: out = (relu(Q) @ attn_inter) / (relu(Q) . ksum)
__global__ void __launch_bounds__(256)
la_phase2(const float* __restrict__ Q, const float* __restrict__ inter,
          const float* __restrict__ ksum, float* __restrict__ out)
{
  __shared__ __align__(16) _Float16 mT[64][72];  // mT[e][d] = attn_inter[d][e] (f16)
  __shared__ float ks[64];

  const int bh   = blockIdx.x;
  const int tid  = threadIdx.x;
  const int wave = tid >> 5;
  const int lane = tid & 31;

  const float* ip = inter + (size_t)bh * 4096;
  #pragma unroll
  for (int it = 0; it < 4; ++it) {
    const int i = (tid + it * 256) * 4;          // element index, multiple of 4
    const float4 w = *(const float4*)(ip + i);
    const int d = i >> 6;
    const int e = i & 63;
    mT[e + 0][d] = (_Float16)w.x;
    mT[e + 1][d] = (_Float16)w.y;
    mT[e + 2][d] = (_Float16)w.z;
    mT[e + 3][d] = (_Float16)w.w;
  }
  if (tid < 64) ks[tid] = ksum[bh * 64 + tid];
  __syncthreads();

  const int row0 = blockIdx.y * ROWS + wave * 16;
  const int m  = lane & 15;
  const int kb = (lane < 16) ? 0 : 8;            // A k-base
  const float* qrow = Q + ((size_t)bh * S_ + row0 + m) * D_;

  // Build A operands straight from global (relu + f16 cvt), fuse fp32 norm dot
  float np = 0.f;
  V16U a0, a1;
  #pragma unroll
  for (int half = 0; half < 2; ++half) {
    #pragma unroll
    for (int g = 0; g < 2; ++g) {
      const int d0 = half * 32 + kb + g * 16;
      const float4 f0 = *(const float4*)(qrow + d0);
      const float4 f1 = *(const float4*)(qrow + d0 + 4);
      const float vals[8] = {f0.x, f0.y, f0.z, f0.w, f1.x, f1.y, f1.z, f1.w};
      #pragma unroll
      for (int j = 0; j < 8; ++j) {
        const float q = fmaxf(vals[j], 0.f);
        np += q * ks[d0 + j];
        if (half == 0) a0.v[g * 8 + j] = (_Float16)q;
        else           a1.v[g * 8 + j] = (_Float16)q;
      }
    }
  }
  np += __shfl_xor(np, 16, 32);                  // combine lane-half partials (wave32)
  const float rn = 1.f / np;                     // rn on lane l = 1/norm[row (l&15)]

  const int bkb = (lane < 16) ? 0 : 8;
  v8f c0 = {}, c1 = {}, c2 = {}, c3 = {};
  v8f* cs[4] = {&c0, &c1, &c2, &c3};
  #pragma unroll
  for (int nt = 0; nt < 4; ++nt) {
    const int n = nt * 16 + (lane & 15);
    V16U b0, b1;
    b0.h[0] = *(const v8h*)&mT[n][bkb];          // k = d 0..31
    b0.h[1] = *(const v8h*)&mT[n][bkb + 16];
    b1.h[0] = *(const v8h*)&mT[n][32 + bkb];     // k = d 32..63
    b1.h[1] = *(const v8h*)&mT[n][32 + bkb + 16];
    *cs[nt] = __builtin_amdgcn_wmma_f32_16x16x32_f16(false, a0.v, false, b0.v, (short)0, *cs[nt], false, false);
    *cs[nt] = __builtin_amdgcn_wmma_f32_16x16x32_f16(false, a1.v, false, b1.v, (short)0, *cs[nt], false, false);
  }

  float* op = out + ((size_t)bh * S_ + row0) * D_;
  #pragma unroll
  for (int r = 0; r < 8; ++r) {
    const int row = r + ((lane < 16) ? 0 : 8);
    const float rnr = __shfl(rn, row, 32);       // fetch 1/norm for this C row
    #pragma unroll
    for (int nt = 0; nt < 4; ++nt)
      op[(size_t)row * 64 + nt * 16 + (lane & 15)] = (*cs[nt])[r] * rnr;
  }
}

extern "C" void kernel_launch(void* const* d_in, const int* in_sizes, int n_in,
                              void* d_out, int out_size, void* d_ws, size_t ws_size,
                              hipStream_t stream) {
  (void)in_sizes; (void)n_in; (void)out_size; (void)ws_size;
  const float* Q    = (const float*)d_in[0];
  const float* K    = (const float*)d_in[1];
  const float* V    = (const float*)d_in[2];
  const int*   mask = (const int*)d_in[3];
  float* out   = (float*)d_out;
  float* inter = (float*)d_ws;                   // BH_*64*64 fp32
  float* ksum  = inter + BH_ * 64 * 64;          // BH_*64 fp32

  const int nz = BH_ * 64 * 64 + BH_ * 64;
  zero_ws_kernel<<<(nz + 255) / 256, 256, 0, stream>>>(inter, nz);
  la_phase1<<<dim3(BH_, S_ / CHUNK), 256, 0, stream>>>(K, V, mask, inter, ksum);
  la_phase2<<<dim3(BH_, S_ / ROWS),  256, 0, stream>>>(Q, inter, ksum, out);
}
